// WaveNet_8495445311691
// MI455X (gfx1250) — compile-verified, matrix operand order checked by reading, not run on hardware
//
#include <hip/hip_runtime.h>

typedef __bf16 bf16;
typedef __attribute__((ext_vector_type(16))) __bf16 v16bf;
typedef __attribute__((ext_vector_type(8)))  __bf16 v8bf;
typedef __attribute__((ext_vector_type(8)))  float  v8f;
typedef __attribute__((ext_vector_type(4)))  int    v4i;

#define B_  4
#define C_  512
#define CP_ 256
#define T_  2048
#define BT_ 8192
#define L_  30
#define KSTEP 32
#define TILE_N 128

// ---------------- CDNA5 feature gates ----------------

#if __has_builtin(__builtin_amdgcn_global_load_async_to_lds_b128) && \
    __has_builtin(__builtin_amdgcn_s_wait_asynccnt)
#define USE_ASYNC_LDS 1
#else
#define USE_ASYNC_LDS 0
#endif

typedef __attribute__((address_space(1))) v4i as1_v4i;
typedef __attribute__((address_space(3))) v4i as3_v4i;

__device__ __forceinline__ float fast_tanh(float x) {
#if __has_builtin(__builtin_amdgcn_tanhf)
  return __builtin_amdgcn_tanhf(x);
#elif __has_builtin(__builtin_amdgcn_tanh_f32)
  return __builtin_amdgcn_tanh_f32(x);
#else
  return tanhf(x);
#endif
}

// Copy one 16-byte chunk global -> LDS (async path when available).
__device__ __forceinline__ void stage16(bf16* lds, const bf16* g) {
#if USE_ASYNC_LDS
  __builtin_amdgcn_global_load_async_to_lds_b128(
      (as1_v4i*)(uintptr_t)g,
      (as3_v4i*)(unsigned)(uintptr_t)lds,
      0, 0);
#else
  *(v8bf*)lds = *(const v8bf*)g;
#endif
}

__device__ __forceinline__ void zero16(bf16* lds) {
  v8bf z = {};
  *(v8bf*)lds = z;   // ds_store_b128
}

// Wait for staging traffic, then workgroup barrier.
__device__ __forceinline__ void stage_fence() {
#if USE_ASYNC_LDS
  __builtin_amdgcn_s_wait_asynccnt(0);
#endif
  __syncthreads();
}

// ---------------- WMMA helpers ----------------

__device__ __forceinline__ v8f wmma_bf16(v16bf a, v16bf b, v8f c) {
  // D = A(16x32 bf16) x B(32x16 bf16) + C(16x16 f32)
  return __builtin_amdgcn_wmma_f32_16x16x32_bf16(
      /*neg_a=*/false, a, /*neg_b=*/false, b,
      /*c_mod=*/(short)0, c, /*reuse_a=*/false, /*reuse_b=*/false);
}

__device__ __forceinline__ v16bf cat16(v8bf lo, v8bf hi) {
  union { v16bf v; v8bf h[2]; } u;
  u.h[0] = lo; u.h[1] = hi;
  return u.v;
}

// 16x32 A-fragment from row-major bf16 global memory, row stride S (elements).
// CDNA5 16-bit A layout: lanes 0-15 -> rows 0..15, K {0..7,16..23};
// lanes 16-31 -> same rows, K {8..15,24..31}.
__device__ __forceinline__ v16bf load_frag(const bf16* __restrict__ p, int S) {
  const int lane = threadIdx.x & 31;
  const bf16* q = p + (size_t)(lane & 15) * S + ((lane >> 4) << 3);
  v8bf a = *(const v8bf*)q;
  v8bf b = *(const v8bf*)(q + 16);
  return cat16(a, b);
}

// 32x16 B-fragment from an LDS tile stored [row][KSTEP] (row = timestep).
__device__ __forceinline__ v16bf lds_frag(const bf16* base) {
  const int lane = threadIdx.x & 31;
  const bf16* q = base + (lane & 15) * KSTEP + ((lane >> 4) << 3);
  v8bf a = *(const v8bf*)q;        // ds_load_b128
  v8bf b = *(const v8bf*)(q + 16); // ds_load_b128
  return cat16(a, b);
}

// ---------------- prep kernels ----------------

__global__ void conv_split_kernel(const float* __restrict__ src,
                                  bf16* __restrict__ w0, bf16* __restrict__ w1,
                                  long n) {
  long i = (long)blockIdx.x * blockDim.x + threadIdx.x;
  if (i < n) {
    w0[i] = (bf16)src[2 * i];
    w1[i] = (bf16)src[2 * i + 1];
  }
}

__global__ void cast_bf_kernel(const float* __restrict__ s, bf16* __restrict__ d, long n) {
  long i = (long)blockIdx.x * blockDim.x + threadIdx.x;
  if (i < n) d[i] = (bf16)s[i];
}

// x [B,C,T] fp32 -> time-major Xf32 [BT][C] + bf16 copy
__global__ void prep_x_kernel(const float* __restrict__ x,
                              float* __restrict__ xf, bf16* __restrict__ xb) {
  long i = (long)blockIdx.x * blockDim.x + threadIdx.x;
  if (i >= (long)B_ * C_ * T_) return;
  int t = (int)(i % T_);
  long r = i / T_;
  int c = (int)(r % C_);
  int b = (int)(r / C_);
  float v = x[i];
  size_t n = (size_t)b * T_ + t;
  xf[n * C_ + c] = v;
  xb[n * C_ + c] = (bf16)v;
}

// pose [B,CP,T] fp32 -> time-major bf16 [BT][CP]
__global__ void prep_pose_kernel(const float* __restrict__ p, bf16* __restrict__ pb) {
  long i = (long)blockIdx.x * blockDim.x + threadIdx.x;
  if (i >= (long)B_ * CP_ * T_) return;
  int t = (int)(i % T_);
  long r = i / T_;
  int c = (int)(r % CP_);
  int b = (int)(r / CP_);
  pb[((size_t)b * T_ + t) * CP_ + c] = (bf16)p[i];
}

// time-major fp32 [BT][C] -> d_out x_final [B,C,T]
__global__ void final_x_kernel(const float* __restrict__ xf, float* __restrict__ out) {
  long i = (long)blockIdx.x * blockDim.x + threadIdx.x;
  if (i >= (long)B_ * C_ * T_) return;
  int t = (int)(i % T_);
  long r = i / T_;
  int c = (int)(r % C_);
  int b = (int)(r / C_);
  out[i] = xf[((size_t)b * T_ + t) * C_ + c];
}

// ---------------- layer kernels ----------------
// Grid: (BT/128, C/64), 256 threads = 8 wave32 (4 M x 2 N); wave tile 16 x 64.
// Activation tiles (B operands) are staged through LDS once per workgroup
// per K-step; weights (A operands) stream from L2 directly.

__global__ __launch_bounds__(256) void wavenet_gate(
    const bf16* __restrict__ W0, const bf16* __restrict__ W1,
    const bf16* __restrict__ CW,
    const float* __restrict__ convb, const float* __restrict__ condb,
    const bf16* __restrict__ Xbf, const bf16* __restrict__ Pbf,
    bf16* __restrict__ Zbf, int d) {
  __shared__ bf16 ldsX[TILE_N * KSTEP];  // x[t] tile      (8 KB)
  __shared__ bf16 ldsS[TILE_N * KSTEP];  // x[t-d] tile    (8 KB)

  const int tid  = threadIdx.x;
  const int w    = tid >> 5;
  const int wm   = w >> 1;
  const int wn   = w & 1;
  const int m0   = blockIdx.y * 64 + wm * 16;    // z-channel rows
  const int nblk = blockIdx.x * TILE_N;          // first timestep of WG tile
  const int nw   = wn * 64;                      // wave column offset in tile
  const int lane = tid & 31;
  const int r16  = lane & 15;
  const int hi   = lane >> 4;

  v8f zero = {0.f, 0.f, 0.f, 0.f, 0.f, 0.f, 0.f, 0.f};
  v8f acct[4], accb[4];
#pragma unroll
  for (int j = 0; j < 4; ++j) { acct[j] = zero; accb[j] = zero; }

  // ---- y = W0 @ x[t-d] + W1 @ x[t] ----
  for (int k = 0; k < C_; k += KSTEP) {
    __syncthreads();  // previous step's LDS reads complete
#pragma unroll
    for (int c = 0; c < 2; ++c) {
      const int chunk = tid * 2 + c;       // 0..511
      const int row   = chunk >> 2;        // 0..127
      const int sub   = (chunk & 3) << 3;  // channel sub-offset {0,8,16,24}
      const int n     = nblk + row;
      stage16(&ldsX[row * KSTEP + sub], Xbf + (size_t)n * C_ + k + sub);
      const int t = n & (T_ - 1);
      if (t >= d) {
        stage16(&ldsS[row * KSTEP + sub], Xbf + (size_t)(n - d) * C_ + k + sub);
      } else {
        zero16(&ldsS[row * KSTEP + sub]);  // causal zero padding
      }
    }
    // A fragments (weights) overlap the staging traffic
    v16bf a0t = load_frag(W0 + (size_t)m0 * C_ + k, C_);
    v16bf a0b = load_frag(W0 + (size_t)(m0 + C_) * C_ + k, C_);
    v16bf a1t = load_frag(W1 + (size_t)m0 * C_ + k, C_);
    v16bf a1b = load_frag(W1 + (size_t)(m0 + C_) * C_ + k, C_);
    if (k + KSTEP < C_) {
      __builtin_prefetch(W0 + (size_t)m0 * C_ + k + KSTEP, 0, 1);
      __builtin_prefetch(W1 + (size_t)m0 * C_ + k + KSTEP, 0, 1);
    }
    stage_fence();
#pragma unroll
    for (int j = 0; j < 4; ++j) {
      const int rloc = nw + 16 * j;
      v16bf bs = lds_frag(&ldsS[rloc * KSTEP]);
      v16bf bx = lds_frag(&ldsX[rloc * KSTEP]);
      acct[j] = wmma_bf16(a0t, bs, acct[j]);
      acct[j] = wmma_bf16(a1t, bx, acct[j]);
      accb[j] = wmma_bf16(a0b, bs, accb[j]);
      accb[j] = wmma_bf16(a1b, bx, accb[j]);
    }
  }

  // ---- y += cond_w @ pose ----
  for (int k = 0; k < CP_; k += KSTEP) {
    __syncthreads();
#pragma unroll
    for (int c = 0; c < 2; ++c) {
      const int chunk = tid * 2 + c;
      const int row   = chunk >> 2;
      const int sub   = (chunk & 3) << 3;
      stage16(&ldsX[row * KSTEP + sub],
              Pbf + (size_t)(nblk + row) * CP_ + k + sub);
    }
    v16bf act = load_frag(CW + (size_t)m0 * CP_ + k, CP_);
    v16bf acb = load_frag(CW + (size_t)(m0 + C_) * CP_ + k, CP_);
    stage_fence();
#pragma unroll
    for (int j = 0; j < 4; ++j) {
      v16bf bp = lds_frag(&ldsX[(nw + 16 * j) * KSTEP]);
      acct[j] = wmma_bf16(act, bp, acct[j]);
      accb[j] = wmma_bf16(acb, bp, accb[j]);
    }
  }

  // ---- gated activation, write Z (bf16, time-major) ----
  float bt[8], bb[8];
#pragma unroll
  for (int r = 0; r < 8; ++r) {
    const int m = m0 + r + (hi << 3);
    bt[r] = convb[m] + condb[m];
    bb[r] = convb[m + C_] + condb[m + C_];
  }
#pragma unroll
  for (int j = 0; j < 4; ++j) {
    const int n = nblk + nw + 16 * j + r16;
#pragma unroll
    for (int r = 0; r < 8; ++r) {
      const int m = m0 + r + (hi << 3);
      float a = acct[j][r] + bt[r];
      float g = accb[j][r] + bb[r];
      float z = fast_tanh(a) * (1.0f / (1.0f + __expf(-g)));
      Zbf[(size_t)n * C_ + m] = (bf16)z;
    }
  }
}

__global__ __launch_bounds__(256) void wavenet_out(
    const bf16* __restrict__ SW, const bf16* __restrict__ RW,
    const float* __restrict__ skipb, const float* __restrict__ resb,
    const bf16* __restrict__ Zbf, const float* __restrict__ Xin,
    float* __restrict__ Xout, bf16* __restrict__ Xbf,
    float* __restrict__ SkipOut) {
  __shared__ bf16 ldsZ[TILE_N * KSTEP];  // z tile (8 KB)

  const int tid  = threadIdx.x;
  const int w    = tid >> 5;
  const int wm   = w >> 1;
  const int wn   = w & 1;
  const int m0   = blockIdx.y * 64 + wm * 16;
  const int nblk = blockIdx.x * TILE_N;
  const int nw   = wn * 64;
  const int lane = tid & 31;
  const int r16  = lane & 15;
  const int hi   = lane >> 4;

  v8f zero = {0.f, 0.f, 0.f, 0.f, 0.f, 0.f, 0.f, 0.f};
  v8f accS[4], accR[4];
#pragma unroll
  for (int j = 0; j < 4; ++j) { accS[j] = zero; accR[j] = zero; }

  for (int k = 0; k < C_; k += KSTEP) {
    __syncthreads();
#pragma unroll
    for (int c = 0; c < 2; ++c) {
      const int chunk = tid * 2 + c;
      const int row   = chunk >> 2;
      const int sub   = (chunk & 3) << 3;
      stage16(&ldsZ[row * KSTEP + sub],
              Zbf + (size_t)(nblk + row) * C_ + k + sub);
    }
    v16bf aS = load_frag(SW + (size_t)m0 * C_ + k, C_);
    v16bf aR = load_frag(RW + (size_t)m0 * C_ + k, C_);
    if (k + KSTEP < C_) {
      __builtin_prefetch(SW + (size_t)m0 * C_ + k + KSTEP, 0, 1);
      __builtin_prefetch(RW + (size_t)m0 * C_ + k + KSTEP, 0, 1);
    }
    stage_fence();
#pragma unroll
    for (int j = 0; j < 4; ++j) {
      v16bf bz = lds_frag(&ldsZ[(nw + 16 * j) * KSTEP]);
      accS[j] = wmma_bf16(aS, bz, accS[j]);
      accR[j] = wmma_bf16(aR, bz, accR[j]);
    }
  }

#pragma unroll
  for (int j = 0; j < 4; ++j) {
    const int n = nblk + nw + 16 * j + r16;
    const int b = n >> 11;            // T = 2048
    const int t = n & (T_ - 1);
#pragma unroll
    for (int r = 0; r < 8; ++r) {
      const int m = m0 + r + (hi << 3);
      SkipOut[((size_t)b * C_ + m) * T_ + t] = accS[j][r] + skipb[m];
      float xn = accR[j][r] + resb[m] + Xin[(size_t)n * C_ + m];
      Xout[(size_t)n * C_ + m] = xn;
      Xbf[(size_t)n * C_ + m]  = (bf16)xn;
    }
  }
}

// ---------------- host ----------------

extern "C" void kernel_launch(void* const* d_in, const int* in_sizes, int n_in,
                              void* d_out, int out_size, void* d_ws, size_t ws_size,
                              hipStream_t stream) {
  const float* x      = (const float*)d_in[0];
  const float* pose   = (const float*)d_in[1];
  const float* conv_w = (const float*)d_in[2];
  const float* conv_b = (const float*)d_in[3];
  const float* cond_w = (const float*)d_in[4];
  const float* cond_b = (const float*)d_in[5];
  const float* res_w  = (const float*)d_in[6];
  const float* res_b  = (const float*)d_in[7];
  const float* skip_w = (const float*)d_in[8];
  const float* skip_b = (const float*)d_in[9];

  char* ws = (char*)d_ws;
  size_t off = 0;
  auto take = [&](size_t bytes) -> char* {
    char* p = ws + off;
    off += (bytes + 255) & ~(size_t)255;
    return p;
  };

  const size_t nConvTap = (size_t)L_ * 2 * C_ * C_;       // per-tap weight count
  bf16*  W0  = (bf16*)take(nConvTap * sizeof(bf16));
  bf16*  W1  = (bf16*)take(nConvTap * sizeof(bf16));
  bf16*  CW  = (bf16*)take((size_t)L_ * 2 * C_ * CP_ * sizeof(bf16));
  bf16*  RW  = (bf16*)take((size_t)L_ * C_ * C_ * sizeof(bf16));
  bf16*  SW  = (bf16*)take((size_t)L_ * C_ * C_ * sizeof(bf16));
  bf16*  Pbf = (bf16*)take((size_t)BT_ * CP_ * sizeof(bf16));
  bf16*  Xbf = (bf16*)take((size_t)BT_ * C_ * sizeof(bf16));
  bf16*  Zbf = (bf16*)take((size_t)BT_ * C_ * sizeof(bf16));
  float* Xf0 = (float*)take((size_t)BT_ * C_ * sizeof(float));
  float* Xf1 = (float*)take((size_t)BT_ * C_ * sizeof(float));

  long n;
  n = (long)nConvTap;
  conv_split_kernel<<<(unsigned)((n + 255) / 256), 256, 0, stream>>>(conv_w, W0, W1, n);
  n = (long)L_ * 2 * C_ * CP_;
  cast_bf_kernel<<<(unsigned)((n + 255) / 256), 256, 0, stream>>>(cond_w, CW, n);
  n = (long)L_ * C_ * C_;
  cast_bf_kernel<<<(unsigned)((n + 255) / 256), 256, 0, stream>>>(res_w, RW, n);
  cast_bf_kernel<<<(unsigned)((n + 255) / 256), 256, 0, stream>>>(skip_w, SW, n);
  n = (long)B_ * C_ * T_;
  prep_x_kernel<<<(unsigned)((n + 255) / 256), 256, 0, stream>>>(x, Xf0, Xbf);
  n = (long)B_ * CP_ * T_;
  prep_pose_kernel<<<(unsigned)((n + 255) / 256), 256, 0, stream>>>(pose, Pbf);

  float* out = (float*)d_out;                 // [x(B,C,T) | skips(L,B,C,T)]
  float* Xin = Xf0;
  float* Xo  = Xf1;
  dim3 grid(BT_ / TILE_N, C_ / 64);

  for (int i = 0; i < L_; ++i) {
    const int d = 1 << (i % 10);
    wavenet_gate<<<grid, 256, 0, stream>>>(
        W0 + (size_t)i * 2 * C_ * C_, W1 + (size_t)i * 2 * C_ * C_,
        CW + (size_t)i * 2 * C_ * CP_,
        conv_b + (size_t)i * 2 * C_, cond_b + (size_t)i * 2 * C_,
        Xbf, Pbf, Zbf, d);
    wavenet_out<<<grid, 256, 0, stream>>>(
        SW + (size_t)i * C_ * C_, RW + (size_t)i * C_ * C_,
        skip_b + (size_t)i * C_, res_b + (size_t)i * C_,
        Zbf, Xin, Xo, Xbf,
        out + (size_t)(i + 1) * B_ * C_ * T_);
    float* tmp = Xin; Xin = Xo; Xo = tmp;
  }

  n = (long)B_ * C_ * T_;
  final_x_kernel<<<(unsigned)((n + 255) / 256), 256, 0, stream>>>(Xin, out);
}
